// CTHead_12756052869725
// MI455X (gfx1250) — matrix operand order repaired and checked by reading, not compiled, not run on hardware
//
#include <hip/hip_runtime.h>
#include <hip/hip_bf16.h>
#include <math.h>

// ---------------- constants ----------------
#define Bsz 16
#define CIN 64
#define Hh  128
#define Wd  128
#define HP  130          // padded H (1-px zero halo)
#define WP  130          // padded W
#define NUM_CLASSES 80
#define KTOP 100
#define SCORE_THRESH 0.2f
#define NMS_THRESH 0.5f
#define RATIO 4.0f
#define HW (Hh*Wd)

typedef _Float16 v16h __attribute__((ext_vector_type(16)));
typedef _Float16 v8h  __attribute__((ext_vector_type(8)));
typedef float    v8f  __attribute__((ext_vector_type(8)));

static __device__ __forceinline__ v16h cat8(v8h lo, v8h hi) {
    return __builtin_shufflevector(lo, hi, 0,1,2,3,4,5,6,7,8,9,10,11,12,13,14,15);
}

// async copy 16B global -> LDS (gfx1250 GLOBAL_LOAD_ASYNC_TO_LDS_B128, ASYNCcnt)
static __device__ __forceinline__ void async_g2l_b128(uint32_t lds_off, const void* gsrc) {
    uint64_t ga = (uint64_t)(size_t)gsrc;
    asm volatile("global_load_async_to_lds_b128 %0, %1, off"
                 :: "v"(lds_off), "v"(ga) : "memory");
}
static __device__ __forceinline__ void wait_async0() {
    asm volatile("s_wait_asynccnt 0x0" ::: "memory");
}

// ---------------- weight repack ----------------
// w1 (OIHW: 64,64,3,3) f32 -> w1t[n][tap*64+cin] f16  (K order = tap-major, cin contiguous)
__global__ void repack_w1_kernel(const float* __restrict__ w1, _Float16* __restrict__ w1t) {
    int i = blockIdx.x * 256 + threadIdx.x;
    if (i >= 64 * 576) return;
    int n = i / 576, k = i % 576;
    int tap = k / 64, cin = k % 64;
    w1t[i] = (_Float16)w1[(n * 64 + cin) * 9 + tap];
}

// w2 (cout,64,1,1) f32 -> w2t[npad][64] f16, zero padded rows
__global__ void repack_w2_kernel(const float* __restrict__ w2, _Float16* __restrict__ w2t,
                                 int cout, int npad) {
    int i = blockIdx.x * 256 + threadIdx.x;
    if (i >= npad * 64) return;
    int co = i / 64, cin = i % 64;
    w2t[i] = (_Float16)(co < cout ? w2[co * 64 + cin] : 0.f);
}

// ---------------- x: NCHW f32 -> padded NHWC f16 (zero halo) ----------------
// one block (256 thr) per padded row (b, hh); hh in [0,130)
__global__ void convert_x_kernel(const float* __restrict__ x, _Float16* __restrict__ xh) {
    const int tid = threadIdx.x;
    const int hh  = blockIdx.x % HP;
    const int b   = blockIdx.x / HP;
    _Float16* row = xh + ((size_t)b * HP + hh) * WP * 64;

    if (hh == 0 || hh == HP - 1) {                 // full zero halo row
        for (int e = tid; e < WP * 64; e += 256) row[e] = (_Float16)0.f;
        return;
    }
    const int h = hh - 1;
    // zero the two halo pixels of this row
    for (int e = tid; e < 2 * 64; e += 256) {
        int ww = (e < 64) ? 0 : (WP - 1);
        row[ww * 64 + (e & 63)] = (_Float16)0.f;
    }
    // interior: coalesced f32 reads along w, f16 writes (pixel-major)
    for (int e = tid; e < 64 * Wd; e += 256) {
        int cin = e >> 7;
        int w   = e & 127;
        const float* p = &x[((size_t)(b * CIN + cin) * Hh + h) * Wd + w];
        __builtin_prefetch(p + Wd, 0, 1);
        row[(w + 1) * 64 + cin] = (_Float16)(*p);
    }
}

// ---------------- fused branch: conv3x3+relu (WMMA) -> conv1x1 (WMMA) -> act ----------------
// grid: Bsz*Hh*(Wd/32) blocks of 256 threads. Block tile: 32 pixels (one row) x 64 mid channels.
// ACT: 0 = sigmoid, 1 = exp
template <int COUT, int NT2, int ACT>
__global__ void branch_kernel(const _Float16* __restrict__ xh,
                              const _Float16* __restrict__ w1t,
                              const float* __restrict__ b1,
                              const _Float16* __restrict__ w2t,
                              const float* __restrict__ b2,
                              float* __restrict__ out) {
    __shared__ __align__(16) _Float16 xs[3 * 34 * 64];   // [row][col(-1..32)][cin]
    __shared__ __align__(16) _Float16 y1s[32 * 64];      // [pixel][mid-channel]

    const int tid  = threadIdx.x;
    const int lane = tid & 31;
    const int wid  = tid >> 5;
    const int g    = lane >> 4;    // lane group 0/1
    const int ln   = lane & 15;

    const int bid  = blockIdx.x;
    const int wblk = bid & 3;
    const int h    = (bid >> 2) & (Hh - 1);
    const int b    = bid >> 9;
    const int w0   = wblk * 32;

    // ---- stage input patch: pure async DMA copy (padded NHWC f16, no bounds checks) ----
    // 3 rows x 34 pixels x 128B = 816 x 16B chunks; LDS layout is exactly linear.
    {
        const uint32_t lds0 = (uint32_t)(size_t)(&xs[0]);
        for (int e = tid; e < 816; e += 256) {
            int chunk = e & 7;        // 16B chunk within one pixel's 128B
            int pc    = e >> 3;       // (r, col)
            int r = pc / 34, col = pc % 34;
            const _Float16* src =
                xh + (((size_t)b * HP + h + r) * WP + w0 + col) * 64 + chunk * 8;
            async_g2l_b128(lds0 + (uint32_t)e * 16, src);
        }
        wait_async0();
    }
    __syncthreads();

    // ---- GEMM1: 16x16 tile per wave, M = 32 pixels (2 tiles), N = 64 ch (4 tiles) ----
    const int mtile = wid >> 2;
    const int ntile = wid & 3;
    const int wl = mtile * 16 + ln;      // local pixel row of A
    const int n1 = ntile * 16 + ln;      // output (mid) channel of B column
    v8f acc = {};
#pragma unroll
    for (int c = 0; c < 2; ++c) {
#pragma unroll
        for (int tap = 0; tap < 9; ++tap) {
            const int dy = tap / 3, dx = tap % 3;
            const _Float16* pa = &xs[(dy * 34 + wl + dx) * 64 + c * 32 + g * 8];
            v16h a = cat8(*(const v8h*)pa, *(const v8h*)(pa + 16));
            const _Float16* pb = &w1t[n1 * 576 + tap * 64 + c * 32 + g * 8];
            v16h bm = cat8(*(const v8h*)pb, *(const v8h*)(pb + 16));
            acc = __builtin_amdgcn_wmma_f32_16x16x32_f16(false, a, false, bm,
                                                         (short)0, acc, false, false);
        }
    }
    {
        const float bias1 = b1[n1];
#pragma unroll
        for (int v = 0; v < 8; ++v) {
            int p = mtile * 16 + v + 8 * g;                  // C layout: M = v + 8*g
            y1s[p * 64 + n1] = (_Float16)fmaxf(acc[v] + bias1, 0.f);
        }
    }
    __syncthreads();

    // ---- GEMM2: 1x1 conv head, K = 64, N = NT2*16 (zero-padded) ----
    for (int t = wid; t < 2 * NT2; t += 8) {
        const int mt = t & 1;
        const int nt = t >> 1;
        const int co = nt * 16 + ln;
        v8f a2 = {};
#pragma unroll
        for (int c = 0; c < 2; ++c) {
            const _Float16* pa = &y1s[(mt * 16 + ln) * 64 + c * 32 + g * 8];
            v16h a = cat8(*(const v8h*)pa, *(const v8h*)(pa + 16));
            const _Float16* pb = &w2t[co * 64 + c * 32 + g * 8];
            v16h bm = cat8(*(const v8h*)pb, *(const v8h*)(pb + 16));
            a2 = __builtin_amdgcn_wmma_f32_16x16x32_f16(false, a, false, bm,
                                                        (short)0, a2, false, false);
        }
        const float bias2 = (co < COUT) ? b2[co] : 0.f;
#pragma unroll
        for (int v = 0; v < 8; ++v) {
            if (co < COUT) {
                int p = mt * 16 + v + 8 * g;
                float val = a2[v] + bias2;
                val = (ACT == 0) ? (1.f / (1.f + __expf(-val))) : __expf(val);
                out[(((size_t)b * Hh + h) * Wd + w0 + p) * COUT + co] = val;
            }
        }
    }
}

// ---------------- peak mask + per-pixel class max/argmax ----------------
__global__ void score_kernel(const float* __restrict__ cls_map,
                             float* __restrict__ scores_all,
                             int* __restrict__ cats_all) {
    int idx = blockIdx.x * 256 + threadIdx.x;
    if (idx >= Bsz * HW) return;
    int b = idx >> 14;            // /16384
    int p = idx & (HW - 1);
    int h = p >> 7, w = p & 127;
    float best = -1.f;
    int bc = 0;
    for (int cat = 0; cat < NUM_CLASSES; ++cat) {
        float center = cls_map[(size_t)idx * NUM_CLASSES + cat];
        float pooled = -INFINITY;
        for (int dy = -1; dy <= 1; ++dy) {
            int hh = h + dy;
            if (hh < 0 || hh >= Hh) continue;
            for (int dx = -1; dx <= 1; ++dx) {
                int ww = w + dx;
                if (ww < 0 || ww >= Wd) continue;
                float v = cls_map[(((size_t)b * Hh + hh) * Wd + ww) * NUM_CLASSES + cat];
                pooled = fmaxf(pooled, v);
            }
        }
        float masked = (center == pooled) ? center : 0.f;
        if (masked > best) { best = masked; bc = cat; }
    }
    scores_all[idx] = best;
    cats_all[idx]   = bc;
}

// ---------------- top-K by iterative argmax (one block per batch) ----------------
__global__ void topk_kernel(float* __restrict__ scores_all,
                            float* __restrict__ tk_scores,
                            int* __restrict__ tk_inds) {
    __shared__ float sv[256];
    __shared__ int   si[256];
    const int b = blockIdx.x, tid = threadIdx.x;
    const int base = b * HW;
    for (int j = 0; j < KTOP; ++j) {
        float best = -INFINITY; int bi = 0;
        for (int p = tid; p < HW; p += 256) {
            float s = scores_all[base + p];
            if (s > best) { best = s; bi = p; }
        }
        sv[tid] = best; si[tid] = bi;
        __syncthreads();
        for (int off = 128; off > 0; off >>= 1) {
            if (tid < off) {
                if (sv[tid + off] > sv[tid] ||
                    (sv[tid + off] == sv[tid] && si[tid + off] < si[tid])) {
                    sv[tid] = sv[tid + off]; si[tid] = si[tid + off];
                }
            }
            __syncthreads();
        }
        if (tid == 0) {
            tk_scores[b * KTOP + j] = sv[0];
            tk_inds[b * KTOP + j]   = si[0];
            scores_all[base + si[0]] = -INFINITY;
        }
        __syncthreads();
    }
}

// ---------------- boxes + sequential NMS (one block per batch) ----------------
__global__ void nms_kernel(const float* __restrict__ tk_scores,
                           const int* __restrict__ tk_inds,
                           const int* __restrict__ cats_all,
                           const float* __restrict__ reg_map,
                           const float* __restrict__ wh_map,
                           float* __restrict__ out) {
    __shared__ float bxS[KTOP * 4];
    __shared__ float areaS[KTOP];
    __shared__ int   keepS[KTOP];
    const int b = blockIdx.x, tid = threadIdx.x;

    float x1 = 0, y1 = 0, x2 = 0, y2 = 0, score = 0;
    int cat = 0;
    if (tid < KTOP) {
        int ind = tk_inds[b * KTOP + tid];
        score   = tk_scores[b * KTOP + tid];
        cat     = cats_all[b * HW + ind];
        float rx = reg_map[((size_t)b * HW + ind) * 2 + 0];
        float ry = reg_map[((size_t)b * HW + ind) * 2 + 1];
        float wx = wh_map [((size_t)b * HW + ind) * 2 + 0];
        float wy = wh_map [((size_t)b * HW + ind) * 2 + 1];
        float gx = (float)(ind & 127), gy = (float)(ind >> 7);
        float cx = gx + rx, cy = gy + ry;
        x1 = (cx - wx * 0.5f) * RATIO; y1 = (cy - wy * 0.5f) * RATIO;
        x2 = (cx + wx * 0.5f) * RATIO; y2 = (cy + wy * 0.5f) * RATIO;
        bxS[tid * 4 + 0] = x1; bxS[tid * 4 + 1] = y1;
        bxS[tid * 4 + 2] = x2; bxS[tid * 4 + 3] = y2;
        areaS[tid] = (x2 - x1) * (y2 - y1);
        keepS[tid] = (score > SCORE_THRESH) ? 1 : 0;
    }
    __syncthreads();

    for (int i = 0; i < KTOP; ++i) {
        int ki = keepS[i];
        if (tid < KTOP && tid > i && ki) {
            float ax1 = bxS[i * 4 + 0], ay1 = bxS[i * 4 + 1];
            float ax2 = bxS[i * 4 + 2], ay2 = bxS[i * 4 + 3];
            float iw = fminf(ax2, x2) - fmaxf(ax1, x1); iw = fmaxf(iw, 0.f);
            float ih = fminf(ay2, y2) - fmaxf(ay1, y1); ih = fmaxf(ih, 0.f);
            float inter = iw * ih;
            float iou = inter / (areaS[i] + areaS[tid] - inter + 1e-9f);
            if (iou > NMS_THRESH) keepS[tid] = 0;
        }
        __syncthreads();
    }

    if (tid < KTOP) {
        const int base_boxes  = 0;
        const int base_cats   = Bsz * KTOP * 4;
        const int base_scores = base_cats + Bsz * KTOP;
        const int base_keep   = base_scores + Bsz * KTOP;
        int r = b * KTOP + tid;
        out[base_boxes + r * 4 + 0] = x1;
        out[base_boxes + r * 4 + 1] = y1;
        out[base_boxes + r * 4 + 2] = x2;
        out[base_boxes + r * 4 + 3] = y2;
        out[base_cats + r]   = (float)cat;
        out[base_scores + r] = score;
        out[base_keep + r]   = (float)keepS[tid];
    }
}

// ---------------- launch ----------------
extern "C" void kernel_launch(void* const* d_in, const int* in_sizes, int n_in,
                              void* d_out, int out_size, void* d_ws, size_t ws_size,
                              hipStream_t stream) {
    const float* x      = (const float*)d_in[0];
    const float* cls_w1 = (const float*)d_in[1];
    const float* cls_b1 = (const float*)d_in[2];
    const float* cls_w2 = (const float*)d_in[3];
    const float* cls_b2 = (const float*)d_in[4];
    const float* reg_w1 = (const float*)d_in[5];
    const float* reg_b1 = (const float*)d_in[6];
    const float* reg_w2 = (const float*)d_in[7];
    const float* reg_b2 = (const float*)d_in[8];
    const float* wh_w1  = (const float*)d_in[9];
    const float* wh_b1  = (const float*)d_in[10];
    const float* wh_w2  = (const float*)d_in[11];
    const float* wh_b2  = (const float*)d_in[12];
    float* out = (float*)d_out;

    char* ws = (char*)d_ws;
    size_t off = 0;
    auto alloc = [&](size_t bytes) -> void* {
        void* p = ws + off;
        off += (bytes + 255) & ~(size_t)255;
        return p;
    };
    _Float16* xh      = (_Float16*)alloc((size_t)Bsz * HP * WP * 64 * sizeof(_Float16));
    _Float16* w1t_cls = (_Float16*)alloc(64 * 576 * sizeof(_Float16));
    _Float16* w1t_reg = (_Float16*)alloc(64 * 576 * sizeof(_Float16));
    _Float16* w1t_wh  = (_Float16*)alloc(64 * 576 * sizeof(_Float16));
    _Float16* w2t_cls = (_Float16*)alloc(80 * 64 * sizeof(_Float16));
    _Float16* w2t_reg = (_Float16*)alloc(16 * 64 * sizeof(_Float16));
    _Float16* w2t_wh  = (_Float16*)alloc(16 * 64 * sizeof(_Float16));
    float* cls_map    = (float*)alloc((size_t)Bsz * HW * NUM_CLASSES * sizeof(float));
    float* reg_map    = (float*)alloc((size_t)Bsz * HW * 2 * sizeof(float));
    float* wh_map     = (float*)alloc((size_t)Bsz * HW * 2 * sizeof(float));
    float* scores_all = (float*)alloc((size_t)Bsz * HW * sizeof(float));
    int*   cats_all   = (int*)alloc((size_t)Bsz * HW * sizeof(int));
    float* tk_scores  = (float*)alloc((size_t)Bsz * KTOP * sizeof(float));
    int*   tk_inds    = (int*)alloc((size_t)Bsz * KTOP * sizeof(int));
    (void)ws_size; (void)in_sizes; (void)n_in; (void)out_size;

    // x -> padded NHWC f16
    convert_x_kernel<<<Bsz * HP, 256, 0, stream>>>(x, xh);

    // weight repack
    repack_w1_kernel<<<(64 * 576 + 255) / 256, 256, 0, stream>>>(cls_w1, w1t_cls);
    repack_w1_kernel<<<(64 * 576 + 255) / 256, 256, 0, stream>>>(reg_w1, w1t_reg);
    repack_w1_kernel<<<(64 * 576 + 255) / 256, 256, 0, stream>>>(wh_w1,  w1t_wh);
    repack_w2_kernel<<<(80 * 64 + 255) / 256, 256, 0, stream>>>(cls_w2, w2t_cls, 80, 80);
    repack_w2_kernel<<<(16 * 64 + 255) / 256, 256, 0, stream>>>(reg_w2, w2t_reg, 2, 16);
    repack_w2_kernel<<<(16 * 64 + 255) / 256, 256, 0, stream>>>(wh_w2,  w2t_wh,  2, 16);

    // fused WMMA branches
    const int nblk = Bsz * Hh * (Wd / 32);   // 8192
    branch_kernel<80, 5, 0><<<nblk, 256, 0, stream>>>(xh, w1t_cls, cls_b1, w2t_cls, cls_b2, cls_map);
    branch_kernel<2, 1, 0><<<nblk, 256, 0, stream>>>(xh, w1t_reg, reg_b1, w2t_reg, reg_b2, reg_map);
    branch_kernel<2, 1, 1><<<nblk, 256, 0, stream>>>(xh, w1t_wh,  wh_b1,  w2t_wh,  wh_b2,  wh_map);

    // peak mask + class max/argmax
    score_kernel<<<(Bsz * HW + 255) / 256, 256, 0, stream>>>(cls_map, scores_all, cats_all);

    // top-K per batch
    topk_kernel<<<Bsz, 256, 0, stream>>>(scores_all, tk_scores, tk_inds);

    // boxes + NMS + output
    nms_kernel<<<Bsz, 128, 0, stream>>>(tk_scores, tk_inds, cats_all, reg_map, wh_map, out);
}
